// GCN_52046413693566
// MI455X (gfx1250) — matrix operand order repaired and checked by reading, not compiled, not run on hardware
//
#include <hip/hip_runtime.h>

typedef __bf16 bf16;
typedef __attribute__((ext_vector_type(16))) __bf16 v16bf;
typedef __attribute__((ext_vector_type(8)))  __bf16 v8bf;
typedef __attribute__((ext_vector_type(4)))  __bf16 v4bf;
typedef __attribute__((ext_vector_type(8)))  float  v8f;

#define TILE_M 128
#define TILE_N 256
#define TILE_K 32
#define LDSS   40   // LDS row stride in bf16 (80B: 16B-aligned, bank-skewed)

// ---------------- elementwise converters ----------------

__global__ __launch_bounds__(256) void k_mul_cvt_bf16(
    const float* __restrict__ a, const float* __restrict__ b,
    bf16* __restrict__ o, int n)
{
    int i = (blockIdx.x * 256 + threadIdx.x) * 4;
    if (i >= n) return;
    float4 va = *(const float4*)(a + i);
    float4 vb = *(const float4*)(b + i);
    v4bf r;
    r[0] = (bf16)(va.x * vb.x);
    r[1] = (bf16)(va.y * vb.y);
    r[2] = (bf16)(va.z * vb.z);
    r[3] = (bf16)(va.w * vb.w);
    *(v4bf*)(o + i) = r;
}

__global__ __launch_bounds__(256) void k_cvt_bf16(
    const float* __restrict__ a, bf16* __restrict__ o, int n)
{
    int i = (blockIdx.x * 256 + threadIdx.x) * 4;
    if (i >= n) return;
    float4 va = *(const float4*)(a + i);
    v4bf r;
    r[0] = (bf16)va.x;
    r[1] = (bf16)va.y;
    r[2] = (bf16)va.z;
    r[3] = (bf16)va.w;
    *(v4bf*)(o + i) = r;
}

// ---------------- CDNA5 helpers ----------------

// Low 32 bits of a flat shared address == LDS byte offset (ISA 10.2 aperture).
__device__ __forceinline__ unsigned lds_addr32(const void* p) {
    return (unsigned)(unsigned long long)(uintptr_t)p;
}

// GLOBAL_LOAD_ASYNC_TO_LDS_B128: 16B global -> LDS, tracked by ASYNCcnt.
__device__ __forceinline__ void async_copy_b128(unsigned ldsoff, const void* gp) {
    unsigned long long ga = (unsigned long long)(uintptr_t)gp;
    asm volatile("global_load_async_to_lds_b128 %0, %1, off"
                 :: "v"(ldsoff), "v"(ga) : "memory");
}

__device__ __forceinline__ void wait_asynccnt0() {
    asm volatile("s_wait_asynccnt 0" ::: "memory");
}

// ---------------- bf16 WMMA GEMM, double-buffered, async A staging ----------
// C[M,N] = A[M,K] @ B[K,N], row-major. A unbatched, B/C batched by blockIdx.z.
// 256 threads = 8 waves (wave32). Block tile 128x256; wave tile 64x64 (4x4 WMMA).

template<bool RELU, bool OUTF32>
__global__ __launch_bounds__(256)
void gemm_bf16_wmma(const bf16* __restrict__ A,
                    const bf16* __restrict__ B,
                    bf16*  __restrict__ Cb,
                    float* __restrict__ Cf,
                    int M, int N, int K,
                    long long strideB, long long strideC)
{
    __shared__ __align__(16) bf16 As[2][TILE_M * LDSS];   // [m][k], row-major
    __shared__ __align__(16) bf16 Bs[2][TILE_N * LDSS];   // [n][k], transposed

    const int tid  = threadIdx.x;
    const int lane = tid & 31;
    const int w    = tid >> 5;
    const int wm   = w & 1;    // 2 wave rows  -> 64 M each
    const int wn   = w >> 1;   // 4 wave cols  -> 64 N each

    const int bm = blockIdx.y * TILE_M;
    const int bn = blockIdx.x * TILE_N;

    const bf16* Bb = B + (long long)blockIdx.z * strideB;
    const long long cbase = (long long)blockIdx.z * strideC;

    v8f acc[4][4];
    #pragma unroll
    for (int m = 0; m < 4; ++m)
        #pragma unroll
        for (int n = 0; n < 4; ++n)
            #pragma unroll
            for (int i = 0; i < 8; ++i)
                acc[m][n][i] = 0.0f;

    // ---- A staging: 128x32 tile = 512 x 16B chunks, async global->LDS ----
    auto stageA = [&](int buf, int k0) {
        #pragma unroll
        for (int s = 0; s < 2; ++s) {
            int c   = tid + s * 256;
            int row = c >> 2;
            int col = (c & 3) << 3;
            async_copy_b128(lds_addr32(&As[buf][row * LDSS + col]),
                            A + (long long)(bm + row) * K + (k0 + col));
        }
    };

    // ---- B staging: 32x256 tile via VGPRs (needs transpose scatter) ----
    v8bf breg[4];
    auto loadB = [&](int k0) {
        #pragma unroll
        for (int s = 0; s < 4; ++s) {
            int c   = tid + s * 256;
            int kr  = c >> 5;              // 0..31
            int col = (c & 31) << 3;       // 0..248
            breg[s] = *(const v8bf*)(Bb + (long long)(k0 + kr) * N + (bn + col));
        }
    };
    auto scatterB = [&](int buf) {
        #pragma unroll
        for (int s = 0; s < 4; ++s) {
            int c   = tid + s * 256;
            int kr  = c >> 5;
            int col = (c & 31) << 3;
            #pragma unroll
            for (int i = 0; i < 8; ++i)
                Bs[buf][(col + i) * LDSS + kr] = breg[s][i];
        }
    };

    // prologue: fill buffer 0
    stageA(0, 0);
    loadB(0);
    scatterB(0);
    wait_asynccnt0();
    __syncthreads();

    const int nsteps = K / TILE_K;
    for (int step = 0; step < nsteps; ++step) {
        const int buf = step & 1;
        const bool has_next = (step + 1) < nsteps;
        if (has_next) {
            stageA(buf ^ 1, (step + 1) * TILE_K);   // async, overlaps compute
            loadB((step + 1) * TILE_K);             // loadcnt waits after WMMAs
        }

        // ---- fragments (ISA 7.12.2 layouts) + 16 WMMAs ----
        const int r  = lane & 15;   // A: row M / B: col N
        const int kg = lane >> 4;   // K half-group select
        v16bf af[4];
        #pragma unroll
        for (int m = 0; m < 4; ++m) {
            const bf16* p = &As[buf][(wm * 64 + m * 16 + r) * LDSS + kg * 8];
            v8bf lo = *(const v8bf*)p;          // K = kg*8 .. +7
            v8bf hi = *(const v8bf*)(p + 16);   // K = kg*8+16 .. +23
            af[m] = __builtin_shufflevector(lo, hi,
                    0,1,2,3,4,5,6,7,8,9,10,11,12,13,14,15);
        }
        v16bf bfr[4];
        #pragma unroll
        for (int n = 0; n < 4; ++n) {
            const bf16* p = &Bs[buf][(wn * 64 + n * 16 + r) * LDSS + kg * 8];
            v8bf lo = *(const v8bf*)p;          // K = kg*8 .. +7
            v8bf hi = *(const v8bf*)(p + 16);   // K = kg*8+16 .. +23
            bfr[n] = __builtin_shufflevector(lo, hi,
                    0,1,2,3,4,5,6,7,8,9,10,11,12,13,14,15);
        }
        #pragma unroll
        for (int m = 0; m < 4; ++m)
            #pragma unroll
            for (int n = 0; n < 4; ++n)
                acc[m][n] = __builtin_amdgcn_wmma_f32_16x16x32_bf16(
                    false, af[m], false, bfr[n], (short)0, acc[m][n],
                    false, false);

        if (has_next) {
            scatterB(buf ^ 1);   // safe: buf^1 last read in step-1, barrier passed
            wait_asynccnt0();    // publish async A tile
        }
        __syncthreads();
    }

    // ---- epilogue: lane%16 -> N col, (lane/16)*8 + vgpr -> M row ----
    const int cn = lane & 15;
    const int rg = lane >> 4;
    #pragma unroll
    for (int m = 0; m < 4; ++m) {
        #pragma unroll
        for (int n = 0; n < 4; ++n) {
            const int col  = bn + wn * 64 + n * 16 + cn;
            const int row0 = bm + wm * 64 + m * 16 + rg * 8;
            #pragma unroll
            for (int i = 0; i < 8; ++i) {
                float v = acc[m][n][i];
                if (RELU) v = v > 0.0f ? v : 0.0f;
                long long idx = cbase + (long long)(row0 + i) * N + col;
                if (OUTF32) Cf[idx] = v;
                else        Cb[idx] = (bf16)v;
            }
        }
    }
}

// ---------------- launcher ----------------

extern "C" void kernel_launch(void* const* d_in, const int* in_sizes, int n_in,
                              void* d_out, int out_size, void* d_ws, size_t ws_size,
                              hipStream_t stream)
{
    const float* x   = (const float*)d_in[0];
    const float* adj = (const float*)d_in[1];
    const float* S   = (const float*)d_in[2];
    const float* W1  = (const float*)d_in[3];
    const float* W2  = (const float*)d_in[4];
    const float* W3  = (const float*)d_in[5];
    float* out = (float*)d_out;

    const int N  = 4096;
    const int NF = 256, NH = 512, NH2 = 1024, NC = 256;
    const int Bt = 4;
    const int BN = Bt * N;   // 16384

    // workspace layout (reused ping-pong regions P/Q)
    char* ws = (char*)d_ws;
    size_t off = 0;
    auto alloc = [&](size_t bytes) {
        char* p = ws + off;
        off += (bytes + 255) & ~(size_t)255;
        return p;
    };
    bf16* adjw = (bf16*)alloc((size_t)N * N * 2);            // 32 MB, persistent
    bf16* w1b  = (bf16*)alloc((size_t)NF * NH * 2);
    bf16* w2b  = (bf16*)alloc((size_t)NH * NH2 * 2);
    bf16* w3b  = (bf16*)alloc((size_t)NH2 * NC * 2);
    char* P    = alloc((size_t)BN * NH2 * 2);                // 32 MB
    char* Q    = alloc((size_t)BN * NH2 * 2);                // 32 MB

    bf16* xb = (bf16*)Q;                                     // 8 MB
    bf16* s1 = (bf16*)(Q + (size_t)BN * NF * 2);             // 16 MB (after xb)
    bf16* h1 = (bf16*)P;
    bf16* s2 = (bf16*)Q;                                     // overwrites xb/s1 (dead)
    bf16* h2 = (bf16*)P;                                     // overwrites h1 (dead)
    bf16* s3 = (bf16*)Q;                                     // overwrites s2 (dead)

    dim3 blk(256);
    auto cgrid = [](int n) { return dim3((unsigned)((n / 4 + 255) / 256)); };

    // weighting = adj * S -> bf16 ; convert x and weights to bf16
    k_mul_cvt_bf16<<<cgrid(N * N), blk, 0, stream>>>(adj, S, adjw, N * N);
    k_cvt_bf16<<<cgrid(BN * NF),  blk, 0, stream>>>(x,  xb,  BN * NF);
    k_cvt_bf16<<<cgrid(NF * NH),  blk, 0, stream>>>(W1, w1b, NF * NH);
    k_cvt_bf16<<<cgrid(NH * NH2), blk, 0, stream>>>(W2, w2b, NH * NH2);
    k_cvt_bf16<<<cgrid(NH2 * NC), blk, 0, stream>>>(W3, w3b, NH2 * NC);

    // layer 1: s1 = x @ W1 ; h1 = relu(adjw @ s1)
    gemm_bf16_wmma<false, false><<<dim3(NH / TILE_N, BN / TILE_M, 1), blk, 0, stream>>>(
        xb, w1b, s1, nullptr, BN, NH, NF, 0, 0);
    gemm_bf16_wmma<true, false><<<dim3(NH / TILE_N, N / TILE_M, Bt), blk, 0, stream>>>(
        adjw, s1, h1, nullptr, N, NH, N, (long long)N * NH, (long long)N * NH);

    // layer 2: s2 = h1 @ W2 ; h2 = relu(adjw @ s2)
    gemm_bf16_wmma<false, false><<<dim3(NH2 / TILE_N, BN / TILE_M, 1), blk, 0, stream>>>(
        h1, w2b, s2, nullptr, BN, NH2, NH, 0, 0);
    gemm_bf16_wmma<true, false><<<dim3(NH2 / TILE_N, N / TILE_M, Bt), blk, 0, stream>>>(
        adjw, s2, h2, nullptr, N, NH2, N, (long long)N * NH2, (long long)N * NH2);

    // layer 3: s3 = h2 @ W3 ; out = adjw @ s3 (f32)
    gemm_bf16_wmma<false, false><<<dim3(NC / TILE_N, BN / TILE_M, 1), blk, 0, stream>>>(
        h2, w3b, s3, nullptr, BN, NC, NH2, 0, 0);
    gemm_bf16_wmma<false, true><<<dim3(NC / TILE_N, N / TILE_M, Bt), blk, 0, stream>>>(
        adjw, s3, nullptr, out, N, NC, N, (long long)N * NC, (long long)N * NC);
}
